// OpenProtModel_42923903156866
// MI455X (gfx1250) — matrix operand order repaired and checked by reading, not compiled, not run on hardware
//
#include <hip/hip_runtime.h>

// ---------------- model constants (reference: DIM=1024, HEADS=16, PD=128, FF=4096, B=1, L=384) ----
#define C_L    384
#define C_DIM  1024
#define C_H    16
#define C_PD   128
#define C_FF   4096

typedef __attribute__((ext_vector_type(16))) _Float16 v16h;
typedef __attribute__((ext_vector_type(2)))  _Float16 v2h;
typedef __attribute__((ext_vector_type(8)))  float    v8f;

// =====================================================================================
// Generic batched WMMA GEMM:
//   C[z][M,N] = act( A[z][M,K] @ op(B[z]) + bias ) (+ add)
//   op(B) = B[K,N] (transB=0) or B[N,K] (transB=1, dot of rows)
// fp32 in/out, f16 WMMA + f32 accumulation.
// Block tile 64x64 (128 threads = 4 waves), each wave computes 2x2 16x16 tiles (4 wmma
// per K-step). Double-buffered LDS + software pipeline: next K-tile is prefetched into
// registers (global_load_b128) while wmma consumes the current LDS buffer; one
// __syncthreads per K-step. Edge tiles use per-element guarded loads into the same
// register block (no divergent loops).
// =====================================================================================
__global__ __launch_bounds__(128)
void gemm_wmma_kernel(const float* __restrict__ A, int lda, long sA,
                      const float* __restrict__ B, int ldb, long sB, int transB,
                      const float* __restrict__ bias,
                      const float* __restrict__ add,
                      float* __restrict__ C, int ldc, long sC,
                      int M, int N, int K, float alpha, int act)
{
    // rows padded to 34 halves: keeps v2h (4B) alignment, odd-dword bank spread
    __shared__ __align__(16) _Float16 As[2][64][34];   // [buf][m][k]
    __shared__ __align__(16) _Float16 Bs[2][64][34];   // [buf][n][k]

    A += (long)blockIdx.z * sA;
    B += (long)blockIdx.z * sB;
    C += (long)blockIdx.z * sC;
    if (add) add += (long)blockIdx.z * sC;

    const int m0   = blockIdx.y * 64;
    const int n0   = blockIdx.x * 64;
    const int tid  = threadIdx.x;
    const int wave = tid >> 5;
    const int lane = tid & 31;
    const int l15  = lane & 15;
    const int lh   = lane >> 4;          // 0: lanes 0-15, 1: lanes 16-31
    const int wm2  = (wave >> 1) * 32;   // wave's 32x32 sub-tile
    const int wn2  = (wave & 1) * 32;

    const bool aFullM = (m0 + 64 <= M);
    const bool bFullN = (n0 + 64 <= N);

    v8f acc00 = {0.f,0.f,0.f,0.f,0.f,0.f,0.f,0.f};
    v8f acc01 = acc00, acc10 = acc00, acc11 = acc00;

    float4 ra[4], rb[4];

    // ---- load one 64x32 A-tile + B-tile (as registers), guarded on edges --------
    auto load_tiles = [&](int k0) {
        const bool kFull = (k0 + 32 <= K);
#pragma unroll
        for (int i = 0; i < 4; ++i) {
            int f4  = tid + i * 128;        // 0..511
            int row = f4 >> 3;              // 0..63
            int c4  = (f4 & 7) * 4;         // 0..28
            if (aFullM && kFull) {
                ra[i] = *(const float4*)(A + (long)(m0 + row) * lda + k0 + c4);
            } else {
                float4 v = {0.f, 0.f, 0.f, 0.f};
                int gm = m0 + row;
                if (gm < M) {
                    const float* p = A + (long)gm * lda;
                    v.x = (k0 + c4 + 0 < K) ? p[k0 + c4 + 0] : 0.f;
                    v.y = (k0 + c4 + 1 < K) ? p[k0 + c4 + 1] : 0.f;
                    v.z = (k0 + c4 + 2 < K) ? p[k0 + c4 + 2] : 0.f;
                    v.w = (k0 + c4 + 3 < K) ? p[k0 + c4 + 3] : 0.f;
                }
                ra[i] = v;
            }
        }
        if (transB) {
#pragma unroll
            for (int i = 0; i < 4; ++i) {
                int f4  = tid + i * 128;
                int row = f4 >> 3;          // n 0..63
                int c4  = (f4 & 7) * 4;     // k offset
                if (bFullN && kFull) {
                    rb[i] = *(const float4*)(B + (long)(n0 + row) * ldb + k0 + c4);
                } else {
                    float4 v = {0.f, 0.f, 0.f, 0.f};
                    int gn = n0 + row;
                    if (gn < N) {
                        const float* p = B + (long)gn * ldb;
                        v.x = (k0 + c4 + 0 < K) ? p[k0 + c4 + 0] : 0.f;
                        v.y = (k0 + c4 + 1 < K) ? p[k0 + c4 + 1] : 0.f;
                        v.z = (k0 + c4 + 2 < K) ? p[k0 + c4 + 2] : 0.f;
                        v.w = (k0 + c4 + 3 < K) ? p[k0 + c4 + 3] : 0.f;
                    }
                    rb[i] = v;
                }
            }
        } else {
#pragma unroll
            for (int i = 0; i < 4; ++i) {
                int f4 = tid + i * 128;     // 512 float4 = 32 k-rows x 16 f4
                int kk = f4 >> 4;           // 0..31
                int n4 = (f4 & 15) * 4;     // 0..60
                if (bFullN && kFull) {
                    rb[i] = *(const float4*)(B + (long)(k0 + kk) * ldb + n0 + n4);
                } else {
                    float4 v = {0.f, 0.f, 0.f, 0.f};
                    int gk = k0 + kk;
                    if (gk < K) {
                        const float* p = B + (long)gk * ldb;
                        v.x = (n0 + n4 + 0 < N) ? p[n0 + n4 + 0] : 0.f;
                        v.y = (n0 + n4 + 1 < N) ? p[n0 + n4 + 1] : 0.f;
                        v.z = (n0 + n4 + 2 < N) ? p[n0 + n4 + 2] : 0.f;
                        v.w = (n0 + n4 + 3 < N) ? p[n0 + n4 + 3] : 0.f;
                    }
                    rb[i] = v;
                }
            }
        }
    };

    // ---- convert regs -> f16 and write the LDS buffer ---------------------------
    auto store_tiles = [&](int buf) {
#pragma unroll
        for (int i = 0; i < 4; ++i) {
            int f4  = tid + i * 128;
            int row = f4 >> 3;
            int c4  = (f4 & 7) * 4;
            v2h h0 = { (_Float16)ra[i].x, (_Float16)ra[i].y };
            v2h h1 = { (_Float16)ra[i].z, (_Float16)ra[i].w };
            *(v2h*)&As[buf][row][c4]     = h0;
            *(v2h*)&As[buf][row][c4 + 2] = h1;
        }
        if (transB) {
#pragma unroll
            for (int i = 0; i < 4; ++i) {
                int f4  = tid + i * 128;
                int row = f4 >> 3;
                int c4  = (f4 & 7) * 4;
                v2h h0 = { (_Float16)rb[i].x, (_Float16)rb[i].y };
                v2h h1 = { (_Float16)rb[i].z, (_Float16)rb[i].w };
                *(v2h*)&Bs[buf][row][c4]     = h0;
                *(v2h*)&Bs[buf][row][c4 + 2] = h1;
            }
        } else {
#pragma unroll
            for (int i = 0; i < 4; ++i) {
                int f4 = tid + i * 128;
                int kk = f4 >> 4;
                int n4 = (f4 & 15) * 4;
                Bs[buf][n4 + 0][kk] = (_Float16)rb[i].x;
                Bs[buf][n4 + 1][kk] = (_Float16)rb[i].y;
                Bs[buf][n4 + 2][kk] = (_Float16)rb[i].z;
                Bs[buf][n4 + 3][kk] = (_Float16)rb[i].w;
            }
        }
    };

    // ---- fragments (CDNA5 ISA layouts) + 4 wmma ----------------------------
    //  A 16x32: lane -> M row; elem e: K = lh*8 + (e<8 ? e : e+8) -> v2h pairs
    //  B 32x16: lane -> N col; elem e: K = lh*8 + (e<8 ? e : e+8) -> v2h pairs
    auto compute = [&](int buf) {
        v16h a0, a1, b0, b1;
        const _Float16* Ar0 = &As[buf][wm2 + l15][0];
        const _Float16* Ar1 = &As[buf][wm2 + 16 + l15][0];
        const _Float16* Br0 = &Bs[buf][wn2 + l15][0];
        const _Float16* Br1 = &Bs[buf][wn2 + 16 + l15][0];
#pragma unroll
        for (int p = 0; p < 8; ++p) {
            int ka = lh * 8 + ((p < 4) ? 2 * p : 16 + 2 * (p - 4));
            v2h pa0 = *(const v2h*)(Ar0 + ka);
            v2h pa1 = *(const v2h*)(Ar1 + ka);
            a0[2 * p] = pa0[0]; a0[2 * p + 1] = pa0[1];
            a1[2 * p] = pa1[0]; a1[2 * p + 1] = pa1[1];
            int kb = lh * 8 + ((p < 4) ? 2 * p : 16 + 2 * (p - 4));
            v2h pb0 = *(const v2h*)(Br0 + kb);
            v2h pb1 = *(const v2h*)(Br1 + kb);
            b0[2 * p] = pb0[0]; b0[2 * p + 1] = pb0[1];
            b1[2 * p] = pb1[0]; b1[2 * p + 1] = pb1[1];
        }
        acc00 = __builtin_amdgcn_wmma_f32_16x16x32_f16(false, a0, false, b0, (short)0, acc00, false, false);
        acc01 = __builtin_amdgcn_wmma_f32_16x16x32_f16(false, a0, false, b1, (short)0, acc01, false, false);
        acc10 = __builtin_amdgcn_wmma_f32_16x16x32_f16(false, a1, false, b0, (short)0, acc10, false, false);
        acc11 = __builtin_amdgcn_wmma_f32_16x16x32_f16(false, a1, false, b1, (short)0, acc11, false, false);
    };

    // ---- software-pipelined main loop -------------------------------------------
    const int NK = (K + 31) / 32;
    load_tiles(0);
    store_tiles(0);
    __syncthreads();
    for (int kt = 0; kt < NK; ++kt) {
        const int cur = kt & 1;
        if (kt + 1 < NK) load_tiles((kt + 1) * 32);   // global loads in flight...
        compute(cur);                                  // ...while wmma consumes LDS
        if (kt + 1 < NK) store_tiles(cur ^ 1);
        __syncthreads();
    }

    // ---- store: C/D f32 16x16 layout: VGPR r -> M = r + 8*lh, N = l15 -----------
#pragma unroll
    for (int tm = 0; tm < 2; ++tm) {
#pragma unroll
        for (int tn = 0; tn < 2; ++tn) {
            const v8f& acc = (tm == 0) ? (tn == 0 ? acc00 : acc01)
                                       : (tn == 0 ? acc10 : acc11);
#pragma unroll
            for (int r = 0; r < 8; ++r) {
                int gm = m0 + wm2 + tm * 16 + lh * 8 + r;
                int gn = n0 + wn2 + tn * 16 + l15;
                if (gm < M && gn < N) {
                    float v = acc[r] * alpha;
                    if (bias) v += bias[gn];
                    if (act == 1) v = v > 0.f ? v : 0.f;
                    else if (act == 2) v = 1.f / (1.f + __expf(-v));
                    if (add) v += add[(long)gm * ldc + gn];
                    C[(long)gm * ldc + gn] = v;
                }
            }
        }
    }
}

// ---------------- LayerNorm over last dim C (one block per row) ----------------------
__global__ __launch_bounds__(128)
void layernorm_kernel(const float* __restrict__ in, const float* __restrict__ g,
                      const float* __restrict__ b, float* __restrict__ out, int C)
{
    const long row = blockIdx.x;
    const float* x = in + row * (long)C;
    float* y = out + row * (long)C;
    __shared__ float s1[128], s2[128];
    float sum = 0.f, sq = 0.f;
    for (int c = threadIdx.x; c < C; c += 128) { float v = x[c]; sum += v; sq += v * v; }
    s1[threadIdx.x] = sum; s2[threadIdx.x] = sq;
    __syncthreads();
    for (int off = 64; off > 0; off >>= 1) {
        if (threadIdx.x < off) {
            s1[threadIdx.x] += s1[threadIdx.x + off];
            s2[threadIdx.x] += s2[threadIdx.x + off];
        }
        __syncthreads();
    }
    float mean = s1[0] / (float)C;
    float var  = s2[0] / (float)C - mean * mean;
    float inv  = rsqrtf(var + 1e-5f);
    for (int c = threadIdx.x; c < C; c += 128)
        y[c] = (x[c] - mean) * inv * g[c] + b[c];
}

// ---- fused: s = softmax_k( s*scale + pairbias[q,k,h], masked ), in place ------------
// s layout [H][L][L]; pb layout [L*L][H]; one block per (h,q) row; L = 384 = 3*128
__global__ __launch_bounds__(128)
void attn_softmax_kernel(float* __restrict__ s, const float* __restrict__ pb,
                         const unsigned char* __restrict__ mask, int L, int H, float scale)
{
    int row = blockIdx.x;
    int h = row / L;
    int q = row - h * L;
    float* sr = s + (long)row * L;
    __shared__ float red[128];
    float v[3];
    float mx = -3.0e38f;
#pragma unroll
    for (int i = 0; i < 3; ++i) {
        int k = threadIdx.x + i * 128;
        float val = -1e9f;
        if (k < L) {
            val = sr[k] * scale + pb[((long)q * L + k) * H + h];
            if (!mask[k]) val = -1e9f;
        }
        v[i] = val;
        mx = fmaxf(mx, val);
    }
    red[threadIdx.x] = mx; __syncthreads();
    for (int off = 64; off > 0; off >>= 1) {
        if (threadIdx.x < off) red[threadIdx.x] = fmaxf(red[threadIdx.x], red[threadIdx.x + off]);
        __syncthreads();
    }
    mx = red[0]; __syncthreads();
    float sum = 0.f;
#pragma unroll
    for (int i = 0; i < 3; ++i) { v[i] = __expf(v[i] - mx); sum += v[i]; }
    red[threadIdx.x] = sum; __syncthreads();
    for (int off = 64; off > 0; off >>= 1) {
        if (threadIdx.x < off) red[threadIdx.x] += red[threadIdx.x + off];
        __syncthreads();
    }
    float r = 1.f / red[0];
#pragma unroll
    for (int i = 0; i < 3; ++i) {
        int k = threadIdx.x + i * 128;
        if (k < L) sr[k] = v[i] * r;
    }
}

// ---------------- elementwise helpers ------------------------------------------------
// out = sg * lin * (mask[i] && mask[j]); layout [L,L,C]
__global__ void gate_mask_kernel(float* __restrict__ out, const float* __restrict__ sg,
                                 const float* __restrict__ lin,
                                 const unsigned char* __restrict__ mask,
                                 int L, int C, long n)
{
    long i = blockIdx.x * (long)blockDim.x + threadIdx.x;
    if (i >= n) return;
    long rc = i / C;
    int i1 = (int)(rc / L), j1 = (int)(rc % L);
    float m = (mask[i1] && mask[j1]) ? 1.f : 0.f;
    out[i] = sg[i] * lin[i] * m;
}

// z += t2 * g
__global__ void gated_residual_kernel(float* __restrict__ z, const float* __restrict__ t2,
                                      const float* __restrict__ g, long n)
{
    long i = blockIdx.x * (long)blockDim.x + threadIdx.x;
    if (i >= n) return;
    z[i] += t2[i] * g[i];
}

// SequenceToPair outer: out[i,j,c] = c<64 ? qh[j,c]*kh[i,c] : qh[j,c-64]-kh[i,c-64]
// where s2 = [L,128], qh = s2[:, :64], kh = s2[:, 64:]
__global__ void s2p_build_kernel(float* __restrict__ out, const float* __restrict__ s2,
                                 int L, int C, long n)
{
    long i = blockIdx.x * (long)blockDim.x + threadIdx.x;
    if (i >= n) return;
    long rc = i / C;
    int c  = (int)(i % C);
    int i1 = (int)(rc / L), j1 = (int)(rc % L);
    int h = C / 2;
    const float* sq = s2 + (long)j1 * C;   // row j -> qh
    const float* sk = s2 + (long)i1 * C;   // row i -> kh
    float v;
    if (c < h) v = sq[c] * sk[h + c];
    else       v = sq[c - h] - sk[c];
    out[i] = v;
}

// pack to channel-major: out[c*LL + rA*L + rB] = in[(r1*L + r2)*C + c]
// swap=0: (r1,r2)=(rA,rB) ; swap=1: (r1,r2)=(rB,rA)  -- coalesced writes, reads hit L2
__global__ void pack_cm_kernel(const float* __restrict__ in, float* __restrict__ out,
                               int L, int C, int swap, long n)
{
    long o = blockIdx.x * (long)blockDim.x + threadIdx.x;
    if (o >= n) return;
    long ll = (long)L * L;
    int c   = (int)(o / ll);
    long rem = o - (long)c * ll;
    int rA = (int)(rem / L), rB = (int)(rem % L);
    int r1 = swap ? rB : rA;
    int r2 = swap ? rA : rB;
    out[o] = in[((long)r1 * L + r2) * C + c];
}

// unpack channel-major back: out[r*C + c] = in[c*LL + r]
__global__ void unpack_cm_kernel(const float* __restrict__ in, float* __restrict__ out,
                                 int L, int C, long n)
{
    long o = blockIdx.x * (long)blockDim.x + threadIdx.x;
    if (o >= n) return;
    long r = o / C;
    int  c = (int)(o % C);
    out[o] = in[(long)c * (long)L * L + r];
}

// =====================================================================================
// host-side launchers
// =====================================================================================
static inline void g_gemm(hipStream_t st, const float* A, int lda, long sA,
                          const float* B, int ldb, long sB, int transB,
                          const float* bias, const float* add,
                          float* C, int ldc, long sC,
                          int M, int N, int K, int batch, int act)
{
    dim3 grid((N + 63) / 64, (M + 63) / 64, batch);
    gemm_wmma_kernel<<<grid, 128, 0, st>>>(A, lda, sA, B, ldb, sB, transB,
                                           bias, add, C, ldc, sC, M, N, K, 1.0f, act);
}
static inline void g_ln(hipStream_t st, const float* in, const float* g, const float* b,
                        float* out, long rows, int C)
{
    layernorm_kernel<<<(unsigned)rows, 128, 0, st>>>(in, g, b, out, C);
}
static inline unsigned eb(long n) { return (unsigned)((n + 255) / 256); }

// ---- parameter flattening (jax sorted-dict order), 56 leaves per block --------------
// 0 bo | 1 ff_b1 | 2 ff_b2 | 3 ff_ln_b | 4 ff_ln_g | 5 ff_w1 | 6 ff_w2 | 7 ffn_b
// 8 ffn_g | 9 mha_b | 10 mha_g | 11 pair_b | 12 pair_g | 13 pff_b1 | 14 pff_b2
// 15 pff_ln2_b | 16 pff_ln2_g | 17 pff_ln_b | 18 pff_ln_g | 19 pff_w1 | 20 pff_w2
// 21 s2p_bo | 22 s2p_bp | 23 s2p_ln_b | 24 s2p_ln_g | 25 s2p_wo | 26 s2p_wp
// 27..38 tmi { ag ap bg bp gb gw ln_in_b ln_in_g ln_out_b ln_out_g zb zw }
// 39..50 tmo { same } | 51 wb | 52 wk | 53 wo | 54 wq | 55 wv
#define TRI_AG      0
#define TRI_AP      1
#define TRI_BG      2
#define TRI_BP      3
#define TRI_GB      4
#define TRI_GW      5
#define TRI_LNIN_B  6
#define TRI_LNIN_G  7
#define TRI_LNOUT_B 8
#define TRI_LNOUT_G 9
#define TRI_ZB      10
#define TRI_ZW      11

extern "C" void kernel_launch(void* const* d_in, const int* in_sizes, int n_in,
                              void* d_out, int out_size, void* d_ws, size_t ws_size,
                              hipStream_t stream)
{
    (void)in_sizes; (void)n_in; (void)out_size; (void)ws_size;
    const int L = C_L, D = C_DIM, H = C_H, PD = C_PD, FF = C_FF;
    const long LL = (long)L * L;
    const long LLPD = LL * PD;

    const float* x_in = (const float*)d_in[0];
    const float* z_in = (const float*)d_in[1];
    const unsigned char* mask = (const unsigned char*)d_in[2];

    // outputs (x, z) concatenated in d_out; compute in place there
    float* x = (float*)d_out;
    float* z = x + (long)L * D;
    hipMemcpyAsync(x, x_in, sizeof(float) * (size_t)L * D, hipMemcpyDeviceToDevice, stream);
    hipMemcpyAsync(z, z_in, sizeof(float) * (size_t)LLPD, hipMemcpyDeviceToDevice, stream);

    // workspace carve-up
    char* w = (char*)d_ws;
    auto alloc = [&](size_t bytes) {
        float* p = (float*)w;
        w += (bytes + 255) & ~(size_t)255;
        return p;
    };
    float* B0 = alloc(sizeof(float) * LLPD);   // zn / pair LN
    float* B1 = alloc(sizeof(float) * LLPD);
    float* B2 = alloc(sizeof(float) * LLPD);
    float* B3 = alloc(sizeof(float) * LLPD);
    float* sbuf   = alloc(sizeof(float) * (size_t)H * LL);   // attention scores/probs
    float* pbbuf  = alloc(sizeof(float) * (size_t)LL * H);   // pair bias [LL,H]
    float* hbuf   = alloc(sizeof(float) * (size_t)L * D);
    float* hbuf2  = alloc(sizeof(float) * (size_t)L * D);
    float* qbuf   = alloc(sizeof(float) * (size_t)L * D);
    float* kbuf   = alloc(sizeof(float) * (size_t)L * D);
    float* vbuf   = alloc(sizeof(float) * (size_t)L * D);
    float* obuf   = alloc(sizeof(float) * (size_t)L * D);
    float* hidbuf = alloc(sizeof(float) * (size_t)L * FF);
    float* s2buf  = alloc(sizeof(float) * (size_t)L * PD);

    for (int blk = 0; blk < 2; ++blk) {
        #define PP(i) ((const float*)d_in[3 + blk * 56 + (i)])

        // ============================ pair-biased MHA ============================
        g_ln(stream, z, PP(12), PP(11), B0, LL, PD);                       // zn = LN(z, pair)
        g_gemm(stream, B0, PD, 0, PP(51), H, 0, 0, nullptr, nullptr,
               pbbuf, H, 0, (int)LL, H, PD, 1, 0);                         // pb = zn @ wb
        g_ln(stream, x, PP(10), PP(9), hbuf, L, D);                        // h = LN(x, mha)
        g_gemm(stream, hbuf, D, 0, PP(54), D, 0, 0, nullptr, nullptr,
               qbuf, D, 0, L, D, D, 1, 0);                                 // q = h @ wq
        g_gemm(stream, hbuf, D, 0, PP(52), D, 0, 0, nullptr, nullptr,
               kbuf, D, 0, L, D, D, 1, 0);                                 // k = h @ wk
        g_gemm(stream, hbuf, D, 0, PP(55), D, 0, 0, nullptr, nullptr,
               vbuf, D, 0, L, D, D, 1, 0);                                 // v = h @ wv
        // s[h] = Q_h @ K_h^T  (batched over heads, head stride 64)
        g_gemm(stream, qbuf, D, 64, kbuf, D, 64, 1, nullptr, nullptr,
               sbuf, L, LL, L, L, D / H, H, 0);
        attn_softmax_kernel<<<H * L, 128, 0, stream>>>(sbuf, pbbuf, mask, L, H,
                                                       0.125f /* 1/sqrt(64) */);
        // o[h] = P_h @ V_h
        g_gemm(stream, sbuf, L, LL, vbuf, D, 64, 0, nullptr, nullptr,
               obuf, D, 64, L, D / H, L, H, 0);
        g_gemm(stream, obuf, D, 0, PP(53), D, 0, 0, PP(0), x,
               x, D, 0, L, D, D, 1, 0);                                    // x += o@wo + bo

        // ============================ gated FFN ==================================
        g_ln(stream, x, PP(8), PP(7), hbuf, L, D);                         // LN(x, ffn)
        g_ln(stream, hbuf, PP(4), PP(3), hbuf2, L, D);                     // LN(., ff_ln)
        g_gemm(stream, hbuf2, D, 0, PP(5), FF, 0, 0, PP(1), nullptr,
               hidbuf, FF, 0, L, FF, D, 1, 1);                             // relu(h@w1+b1)
        g_gemm(stream, hidbuf, FF, 0, PP(6), D, 0, 0, PP(2), x,
               x, D, 0, L, D, FF, 1, 0);                                   // x += hid@w2+b2

        // ============================ SequenceToPair =============================
        g_ln(stream, x, PP(24), PP(23), hbuf, L, D);                       // LN(x, s2p)
        g_gemm(stream, hbuf, D, 0, PP(26), PD, 0, 0, PP(22), nullptr,
               s2buf, PD, 0, L, PD, D, 1, 0);                              // s2 = h@wp+bp
        s2p_build_kernel<<<eb(LLPD), 256, 0, stream>>>(B1, s2buf, L, PD, LLPD);
        g_gemm(stream, B1, PD, 0, PP(25), PD, 0, 0, PP(21), z,
               z, PD, 0, (int)LL, PD, PD, 1, 0);                           // z += pa@wo+bo

        // ============================ triangle multiplications ===================
        for (int dir = 0; dir < 2; ++dir) {
            const int T = (dir == 0) ? 39 : 27;     // tmo first (outgoing), then tmi
            const int swap = (dir == 0) ? 0 : 1;    // incoming contracts leading index
            g_ln(stream, z, PP(T + TRI_LNIN_G), PP(T + TRI_LNIN_B), B0, LL, PD); // zn
            // a = sigmoid(zn@ag) * (zn@ap) * m2
            g_gemm(stream, B0, PD, 0, PP(T + TRI_AG), PD, 0, 0, nullptr, nullptr,
                   B1, PD, 0, (int)LL, PD, PD, 1, 2);
            g_gemm(stream, B0, PD, 0, PP(T + TRI_AP), PD, 0, 0, nullptr, nullptr,
                   B2, PD, 0, (int)LL, PD, PD, 1, 0);
            gate_mask_kernel<<<eb(LLPD), 256, 0, stream>>>(B1, B1, B2, mask, L, PD, LLPD);
            // b = sigmoid(zn@bg) * (zn@bp) * m2
            g_gemm(stream, B0, PD, 0, PP(T + TRI_BG), PD, 0, 0, nullptr, nullptr,
                   B2, PD, 0, (int)LL, PD, PD, 1, 2);
            g_gemm(stream, B0, PD, 0, PP(T + TRI_BP), PD, 0, 0, nullptr, nullptr,
                   B3, PD, 0, (int)LL, PD, PD, 1, 0);
            gate_mask_kernel<<<eb(LLPD), 256, 0, stream>>>(B2, B2, B3, mask, L, PD, LLPD);
            // channel-major packing, then per-channel t_c = A_c @ B_c^T (batch = PD)
            pack_cm_kernel<<<eb(LLPD), 256, 0, stream>>>(B1, B3, L, PD, swap, LLPD); // aT
            pack_cm_kernel<<<eb(LLPD), 256, 0, stream>>>(B2, B1, L, PD, swap, LLPD); // bT
            g_gemm(stream, B3, L, LL, B1, L, LL, 1, nullptr, nullptr,
                   B2, L, LL, L, L, L, PD, 0);                             // tT[c,i,j]
            unpack_cm_kernel<<<eb(LLPD), 256, 0, stream>>>(B2, B3, L, PD, LLPD); // t[i,j,c]
            g_ln(stream, B3, PP(T + TRI_LNOUT_G), PP(T + TRI_LNOUT_B), B1, LL, PD);
            g_gemm(stream, B1, PD, 0, PP(T + TRI_ZW), PD, 0, 0, PP(T + TRI_ZB), nullptr,
                   B2, PD, 0, (int)LL, PD, PD, 1, 0);                      // t2 = LN(t)@zw+zb
            g_gemm(stream, B0, PD, 0, PP(T + TRI_GW), PD, 0, 0, PP(T + TRI_GB), nullptr,
                   B3, PD, 0, (int)LL, PD, PD, 1, 2);                      // g = sig(zn@gw+gb)
            gated_residual_kernel<<<eb(LLPD), 256, 0, stream>>>(z, B2, B3, LLPD);
        }

        // ============================ pair FFN ===================================
        g_ln(stream, z, PP(18), PP(17), B0, LL, PD);                       // LN(z, pff_ln)
        g_ln(stream, B0, PP(16), PP(15), B1, LL, PD);                      // LN(., pff_ln2)
        g_gemm(stream, B1, PD, 0, PP(19), PD, 0, 0, PP(13), nullptr,
               B2, PD, 0, (int)LL, PD, PD, 1, 1);                          // relu(h@w1+b1)
        g_gemm(stream, B2, PD, 0, PP(20), PD, 0, 0, PP(14), z,
               z, PD, 0, (int)LL, PD, PD, 1, 0);                           // z += hid@w2+b2

        #undef PP
    }
}
